// CoPE_5789615915332
// MI455X (gfx1250) — matrix operand order repaired and checked by reading, not compiled, not run on hardware
//
#include <hip/hip_runtime.h>

// CoPE (mode=1) for MI455X / gfx1250.
// Pass 1: key_p = key @ w_k (bf16, row-major), logits_int = q @ pos_emb (f32),
//         q -> bf16, all into L2-resident workspace (~25 MB << 192 MB L2).
// Pass 2: fused streaming pass over the 201 MB output: per 16x16 tile of
//         logits = 0.125 * q @ key_p^T via v_wmma_f32_16x16x32_bf16,
//         sigmoid, reversed (suffix) cumsum via 16-lane shuffles + carry,
//         clamp/floor/ceil, gather-interpolate from logits_int held in LDS.

#define S_DIM 1024
#define D_DIM 64
#define NPOS  64
#define BH_COUNT 48
#define QK_SCALE 0.125f

typedef __attribute__((ext_vector_type(16))) __bf16   v16bf;
typedef __attribute__((ext_vector_type(8)))  float    v8f;
typedef __attribute__((ext_vector_type(8)))  unsigned v8u;

static __device__ __forceinline__ unsigned pk_bf16(float a, float b) {
  unsigned ua = __builtin_bit_cast(unsigned, a);
  unsigned ub = __builtin_bit_cast(unsigned, b);
  ua = (ua + 0x7FFFu + ((ua >> 16) & 1u)) >> 16;   // RNE f32 -> bf16
  ub = (ub + 0x7FFFu + ((ub >> 16) & 1u)) >> 16;
  return (ua & 0xFFFFu) | (ub << 16);
}
static __device__ __forceinline__ unsigned short bf16_1(float a) {
  unsigned ua = __builtin_bit_cast(unsigned, a);
  return (unsigned short)((ua + 0x7FFFu + ((ua >> 16) & 1u)) >> 16);
}
static __device__ __forceinline__ v16bf as_bf(v8u u) {
  return __builtin_bit_cast(v16bf, u);
}

// ---------------------------------------------------------------------------
// Pass 1: one wave handles 16 sequence rows of one (b,h) head.
//   - builds bf16 A-fragments (contiguous K order: k = 16*half + s) for q and key rows
//   - key_p tile = key @ w_k     -> bf16 row-major [bh][s][d]  (workspace)
//   - logits_int = q @ pos_emb   -> f32 [bh][s][NPOS]          (workspace)
//   - q bf16 copy (A-layout-friendly row-major)                (workspace)
// ---------------------------------------------------------------------------
__global__ __launch_bounds__(32) void cope_proj_kernel(
    const float* __restrict__ q, const float* __restrict__ k,
    const float* __restrict__ pos_emb, const float* __restrict__ w_k,
    unsigned* __restrict__ qbf, unsigned short* __restrict__ kpbf,
    float* __restrict__ lint)
{
  const int lane = threadIdx.x & 31;
  const int seg  = lane & 15;     // A: M row; B: N column; C: N column
  const int half = lane >> 4;
  const int st   = blockIdx.x & 63;   // S/16 tiles
  const int bh   = blockIdx.x >> 6;
  const int s0   = st * 16;

  const size_t rowOff = ((size_t)bh * S_DIM + s0 + seg) * D_DIM;

  // A fragments (16x32 bf16 each; two K-halves cover K=0..63)
  v8u aq0, aq1, ak0, ak1;
#pragma unroll
  for (int r = 0; r < 8; ++r) {
    const int kb = 2 * r + 16 * half;  // contiguous K order (matches B)
    aq0[r] = pk_bf16(q[rowOff + kb],      q[rowOff + kb + 1]);
    aq1[r] = pk_bf16(q[rowOff + kb + 32], q[rowOff + kb + 33]);
    ak0[r] = pk_bf16(k[rowOff + kb],      k[rowOff + kb + 1]);
    ak1[r] = pk_bf16(k[rowOff + kb + 32], k[rowOff + kb + 33]);
    qbf[(rowOff + kb) >> 1]      = aq0[r];                 // bf16 pair store
    qbf[(rowOff + kb + 32) >> 1] = aq1[r];
  }

  const int kbB = 16 * half;  // B layout: lanes 0-15 -> K 0..15, 16-31 -> K 16..31
#pragma unroll
  for (int nt = 0; nt < 4; ++nt) {
    const int n = nt * 16 + seg;
    v8u bp0, bp1, bw0, bw1;
#pragma unroll
    for (int r = 0; r < 8; ++r) {
      const int k0 = kbB + 2 * r;
      bp0[r] = pk_bf16(pos_emb[(size_t)k0 * NPOS + n],        pos_emb[(size_t)(k0 + 1) * NPOS + n]);
      bp1[r] = pk_bf16(pos_emb[(size_t)(k0 + 32) * NPOS + n], pos_emb[(size_t)(k0 + 33) * NPOS + n]);
      bw0[r] = pk_bf16(w_k[(size_t)k0 * D_DIM + n],           w_k[(size_t)(k0 + 1) * D_DIM + n]);
      bw1[r] = pk_bf16(w_k[(size_t)(k0 + 32) * D_DIM + n],    w_k[(size_t)(k0 + 33) * D_DIM + n]);
    }
    v8f cI = {};
    cI = __builtin_amdgcn_wmma_f32_16x16x32_bf16(false, as_bf(aq0), false, as_bf(bp0), (short)0, cI, false, false);
    cI = __builtin_amdgcn_wmma_f32_16x16x32_bf16(false, as_bf(aq1), false, as_bf(bp1), (short)0, cI, false, false);
    v8f cK = {};
    cK = __builtin_amdgcn_wmma_f32_16x16x32_bf16(false, as_bf(ak0), false, as_bf(bw0), (short)0, cK, false, false);
    cK = __builtin_amdgcn_wmma_f32_16x16x32_bf16(false, as_bf(ak1), false, as_bf(bw1), (short)0, cK, false, false);
#pragma unroll
    for (int r = 0; r < 8; ++r) {  // C layout: VGPR r -> M=r (lanes 0-15) / r+8
      const int m = r + 8 * half;
      const size_t o = ((size_t)bh * S_DIM + s0 + m) * D_DIM + nt * 16 + seg;
      lint[o] = cI[r];
      kpbf[o] = bf16_1(cK[r]);
    }
  }
}

// ---------------------------------------------------------------------------
// Pass 2: block = 4 waves, each wave owns 16 query rows. Walk key tiles
// right-to-left carrying the suffix sum; fused sigmoid/scan/gather/store.
// ---------------------------------------------------------------------------
__global__ __launch_bounds__(128) void cope_main_kernel(
    const unsigned* __restrict__ qbf, const unsigned* __restrict__ kpbf,
    const float* __restrict__ lint, float* __restrict__ out)
{
  __shared__ float sLint[64 * NPOS];   // 16 KB: logits_int for this block's rows
  const int bh = blockIdx.x >> 4;      // 16 row-supertiles (of 64 rows) per head
  const int rt = blockIdx.x & 15;
  const int rowBase0 = rt * 64;

  {
    const float4* src = (const float4*)(lint + ((size_t)bh * S_DIM + rowBase0) * NPOS);
    float4* dst = (float4*)sLint;
    for (int i = threadIdx.x; i < 64 * NPOS / 4; i += 128) dst[i] = src[i];
  }
  __syncthreads();

  const int wave = threadIdx.x >> 5;
  const int lane = threadIdx.x & 31;
  const int seg  = lane & 15;
  const int half = lane >> 4;
  const int rowT = rowBase0 + wave * 16;

  // Resident A fragments for this wave's 16 query rows (K = 0..63)
  const unsigned* qrow = qbf + ((size_t)bh * S_DIM + rowT + seg) * (D_DIM / 2);
  v8u a0u, a1u;
#pragma unroll
  for (int r = 0; r < 8; ++r) {
    const int kb = 2 * r + 16 * half;
    a0u[r] = qrow[kb >> 1];
    a1u[r] = qrow[(kb + 32) >> 1];
  }
  const v16bf a0 = as_bf(a0u), a1 = as_bf(a1u);

  float carry[8];
#pragma unroll
  for (int r = 0; r < 8; ++r) carry[r] = 0.0f;

  for (int j = (S_DIM / 16) - 1; j >= 0; --j) {
    // B fragments: key_p bf16 row-major [s][d] -> two contiguous 32B loads
    const unsigned* kcol = kpbf + ((size_t)bh * S_DIM + j * 16 + seg) * (D_DIM / 2);
    if (j > 0)
      __builtin_prefetch(kpbf + ((size_t)bh * S_DIM + (j - 1) * 16 + seg) * (D_DIM / 2), 0, 1);
    const v8u b0u = *(const v8u*)(kcol + 8 * half);        // K(d) = 0..31
    const v8u b1u = *(const v8u*)(kcol + 16 + 8 * half);   // K(d) = 32..63

    v8f c = {};
    c = __builtin_amdgcn_wmma_f32_16x16x32_bf16(false, a0, false, as_bf(b0u), (short)0, c, false, false);
    c = __builtin_amdgcn_wmma_f32_16x16x32_bf16(false, a1, false, as_bf(b1u), (short)0, c, false, false);

#pragma unroll
    for (int r = 0; r < 8; ++r) {
      const float g = 1.0f / (1.0f + __expf(-(c[r] * QK_SCALE)));   // sigmoid
      // inclusive suffix sum across this row's 16 columns (one 16-lane segment)
      float ssum = g;
#pragma unroll
      for (int off = 1; off < 16; off <<= 1) {
        const float t = __shfl_down(ssum, off, 32);
        ssum += (seg + off < 16) ? t : 0.0f;
      }
      const float tot = __shfl(ssum, lane & 16, 32);   // segment lane 0 = tile-row total
      float pos = ssum + carry[r];
      carry[r] += tot;
      pos = fminf(pos, (float)(NPOS - 1));
      const float pf = floorf(pos);
      const float w  = pos - pf;
      const int ipf  = (int)pf;
      int ipc = ipf + ((w > 0.0f) ? 1 : 0);
      ipc = (ipc > NPOS - 1) ? (NPOS - 1) : ipc;

      const int rloc = wave * 16 + r + 8 * half;
      const float lf = sLint[rloc * NPOS + ipf];
      const float lc = sLint[rloc * NPOS + ipc];
      out[((size_t)bh * S_DIM + rowT + r + 8 * half) * S_DIM + j * 16 + seg] =
          lc * w + lf * (1.0f - w);
    }
  }
}

extern "C" void kernel_launch(void* const* d_in, const int* in_sizes, int n_in,
                              void* d_out, int out_size, void* d_ws, size_t ws_size,
                              hipStream_t stream) {
  const float* q       = (const float*)d_in[0];
  // d_in[1] attn_logits: unused in mode 1
  const float* key     = (const float*)d_in[2];
  // d_in[3] value: unused
  const float* pos_emb = (const float*)d_in[4];
  const float* w_k     = (const float*)d_in[5];
  // d_in[6] is_cope_k == 1 in the reference setup (mode fixed at compile time)

  // Workspace layout (all 256B-aligned):
  //   qbf  : BH*S*D bf16  = 6.29 MB
  //   kpbf : BH*S*D bf16  = 6.29 MB
  //   lint : BH*S*NPOS f32 = 12.58 MB
  const size_t BF_BYTES = (size_t)BH_COUNT * S_DIM * D_DIM * 2;
  char* ws = (char*)d_ws;
  unsigned*       qbf  = (unsigned*)ws;
  unsigned short* kpbf = (unsigned short*)(ws + BF_BYTES);
  float*          lint = (float*)(ws + 2 * BF_BYTES);

  cope_proj_kernel<<<BH_COUNT * (S_DIM / 16), 32, 0, stream>>>(
      q, key, pos_emb, w_k, qbf, kpbf, lint);
  cope_main_kernel<<<BH_COUNT * (S_DIM / 64), 128, 0, stream>>>(
      qbf, (const unsigned*)kpbf, lint, (float*)d_out);
}
